// LSTMPosterior_25451976196387
// MI455X (gfx1250) — matrix operand order repaired and checked by reading, not compiled, not run on hardware
//
#include <hip/hip_runtime.h>
#include <hip/hip_bf16.h>

// ---------------------------------------------------------------------------
// LSTM for MI455X (gfx1250, wave32, WMMA + TDM).
//   B=64, T=512, D=256, H=1024, 4H=4096.
//   K0 prep   : fp32->bf16 conversions (x, W_ih, W_hh), zero h ping-pong+bar.
//   K1 xproj  : x_proj = x @ W_ih^T + b  via v_wmma_f32_16x16x32_bf16 -> fp32.
//   K2 recur  : 32 persistent WGs; each stages its 128x1024 bf16 W_hh slice
//               into LDS via 4 TENSOR_LOAD_TO_LDS ops (hardware padding gives
//               the bank-conflict-free 1032-elem row pitch), then runs the
//               512-step recurrence: 4 gate tiles x 32 bf16 WMMAs per wave,
//               register-resident cell state, one agent-fenced grid barrier
//               per step on ping-pong bf16 h buffers.
// ---------------------------------------------------------------------------

#define BB   64
#define TT   512
#define DD   256
#define HH   1024
#define G4   4096
#define NWG  32          // workgroups in recurrent kernel (one per WGP)
#define KP   1032        // LDS row pitch in bf16 elems (1024 + 2*4 pad)
#define PADH 4           // extra bf16 elems inserted after k=512 within a row

typedef __attribute__((ext_vector_type(16))) __bf16 v16bf;
typedef __attribute__((ext_vector_type(8)))  float  v8f;
typedef __attribute__((ext_vector_type(4)))  unsigned int u32x4;
typedef __attribute__((ext_vector_type(8)))  int  i32x8;
typedef __attribute__((ext_vector_type(4)))  int  i32x4;

union AFrag { v16bf v; unsigned u[8]; };
union BFrag { v16bf v; uint4 q[2]; };

static __device__ __forceinline__ unsigned short f2bf(float f) {
  unsigned u = __float_as_uint(f);
  unsigned r = (u + 0x7FFFu + ((u >> 16) & 1u)) >> 16;   // round-nearest-even
  return (unsigned short)r;
}
static __device__ __forceinline__ float sigmoidf_(float x) {
  return 1.0f / (1.0f + __expf(-x));
}

// A-matrix (16x32 bf16) per-lane dword k-offsets: VGPR w holds
// K = {2w,2w+1} (16 consecutive K), +16 for lanes 16..31.
static __device__ __forceinline__ void load_afrag(AFrag& a,
                                                  const unsigned short* base) {
  a.u[0] = *(const unsigned*)(base + 0);
  a.u[1] = *(const unsigned*)(base + 2);
  a.u[2] = *(const unsigned*)(base + 4);
  a.u[3] = *(const unsigned*)(base + 6);
  a.u[4] = *(const unsigned*)(base + 8);
  a.u[5] = *(const unsigned*)(base + 10);
  a.u[6] = *(const unsigned*)(base + 12);
  a.u[7] = *(const unsigned*)(base + 14);
}

static __device__ __forceinline__ void grid_barrier(unsigned* bar,
                                                    unsigned target) {
  __syncthreads();
  if (threadIdx.x == 0) {
    __builtin_amdgcn_fence(__ATOMIC_RELEASE, "agent");   // global_wb
    __hip_atomic_fetch_add(bar, 1u, __ATOMIC_RELAXED, __HIP_MEMORY_SCOPE_AGENT);
    while (__hip_atomic_load(bar, __ATOMIC_RELAXED,
                             __HIP_MEMORY_SCOPE_AGENT) < target) {
      __builtin_amdgcn_s_sleep(1);
    }
    __builtin_amdgcn_fence(__ATOMIC_ACQUIRE, "agent");   // global_inv
  }
  __syncthreads();
}

// ---------------------------------------------------------------------------
// K0: conversions + zero init (deterministic every launch).
// ---------------------------------------------------------------------------
__global__ void lstm_prep_kernel(const float* __restrict__ x,
                                 const float* __restrict__ wih,
                                 const float* __restrict__ whh,
                                 unsigned short* __restrict__ x_bf,
                                 unsigned short* __restrict__ wih_bf,
                                 unsigned short* __restrict__ whh_bf,
                                 unsigned short* __restrict__ h_bf,
                                 unsigned* __restrict__ bar) {
  size_t tid = (size_t)blockIdx.x * blockDim.x + threadIdx.x;
  size_t stride = (size_t)gridDim.x * blockDim.x;
  const size_t NX  = (size_t)BB * TT * DD;      // 8.4M
  const size_t NWI = (size_t)G4 * DD;           // 1.05M
  const size_t NWH = (size_t)G4 * HH;           // 4.2M
  const size_t NH  = 2u * BB * HH;              // ping-pong h
  for (size_t i = tid; i < NX;  i += stride) x_bf[i]  = f2bf(x[i]);
  for (size_t i = tid; i < NWI; i += stride) wih_bf[i] = f2bf(wih[i]);
  for (size_t i = tid; i < NWH; i += stride) whh_bf[i] = f2bf(whh[i]);
  for (size_t i = tid; i < NH;  i += stride) h_bf[i] = 0;
  if (tid == 0) *bar = 0u;
}

// ---------------------------------------------------------------------------
// K1: x_proj = x @ W_ih^T + b. One 16x16 tile (K=256 -> 8 WMMAs) per wave.
// Grid: 65536 blocks x 256 thr (8 waves) = 524288 tiles = 2048 x 256 tiles.
// ---------------------------------------------------------------------------
__global__ void lstm_xproj_kernel(const unsigned short* __restrict__ x_bf,
                                  const unsigned short* __restrict__ wih_bf,
                                  const float* __restrict__ bias,
                                  float* __restrict__ xp) {
  const int wave = threadIdx.x >> 5;
  const int lane = threadIdx.x & 31;
  const int tile = blockIdx.x * 8 + wave;
  const int tc   = tile & 255;          // col tile (of 4096/16)
  const int tr   = tile >> 8;           // row tile (of 32768/16)
  const int nloc = lane & 15;
  const int hi   = lane >> 4;

  const int arow = tr * 16 + nloc;      // A lane row (M)
  const int bcol = tc * 16 + nloc;      // B lane col (N) == W_ih row

  v8f acc = {0.f, 0.f, 0.f, 0.f, 0.f, 0.f, 0.f, 0.f};
#pragma unroll
  for (int k0 = 0; k0 < DD; k0 += 32) {
    AFrag a;
    load_afrag(a, x_bf + (size_t)arow * DD + k0 + hi * 16);
    BFrag bb;   // B tile: lane holds 16 consecutive K (lanes>=16: +16)
    const uint4* bp =
        (const uint4*)(wih_bf + (size_t)bcol * DD + k0 + hi * 16);
    bb.q[0] = bp[0];
    bb.q[1] = bp[1];
    acc = __builtin_amdgcn_wmma_f32_16x16x32_bf16(
        false, a.v, false, bb.v, (short)0, acc, false, false);
  }
  const float bv = bias[bcol];
#pragma unroll
  for (int v = 0; v < 8; ++v) {
    const int row = tr * 16 + v + hi * 8;        // C/D layout: M = v (+8)
    xp[(size_t)row * G4 + bcol] = acc[v] + bv;
  }
}

// ---------------------------------------------------------------------------
// K2: recurrence. blockIdx.x = g in [0,32): owns h-cols [g*32, g*32+32).
// Wave w: rt = w&3 (batch rows rt*16..), ct = w>>2 (col half).
// LDS layout (matches TDM hardware padding): element (lc, k) at bf16 index
//   lc*KP + k + (k>=512 ? PADH : 0); row pitch KP=1032 -> 4-dword bank
//   rotation per gate-column => conflict-free ds_load_b128 B-fragments.
// ---------------------------------------------------------------------------
__global__ void lstm_recur_kernel(const float* __restrict__ xp,
                                  const unsigned short* __restrict__ whh_bf,
                                  unsigned short* __restrict__ h_bf,
                                  float* __restrict__ out,
                                  unsigned* __restrict__ bar) {
  extern __shared__ unsigned short lds_w[];     // 128 * KP bf16 = 258 KB

  const int tid  = threadIdx.x;
  const int wave = tid >> 5;
  const int lane = tid & 31;
  const int rt   = wave & 3;
  const int ct   = wave >> 2;
  const int g    = blockIdx.x;
  const int nloc = lane & 15;
  const int hi   = lane >> 4;

#if defined(__gfx1250__) && __has_builtin(__builtin_amdgcn_tensor_load_to_lds)
  // --- Tensor Data Mover staging: 4 tiles (one per gate), each 1024 x 32
  // bf16 rows from whh_bf, with D# pad_enable inserting 2 dwords after every
  // 256 dwords => LDS row pitch = 1032 bf16 (KP), mid-row pad at k=512.
  if (wave == 0) {
    const unsigned long long gbase = (unsigned long long)whh_bf;
    const unsigned lbase = (unsigned)(size_t)&lds_w[0];
    // D# group1: data_size=2B(1), pad_enable, pad_interval=256dw(7),
    // pad_amount=2dw(1); tensor_dim0=1024, tensor_dim1=32, tile=1024x32,
    // tensor_dim0_stride=1024.
    const i32x8 g1 = {
        (int)((1u << 16) | (1u << 20) | (7u << 22) | (1u << 25)),
        (int)(1024u << 16),        // tensor_dim0[15:0] in bits [63:48]
        (int)(32u << 16),          // dim0 hi=0 | tensor_dim1[15:0]
        (int)(1024u << 16),        // dim1 hi=0 | tile_dim0
        (int)32,                   // tile_dim1=32, tile_dim2=0
        (int)1024,                 // tensor_dim0_stride[31:0]
        0, 0};
    const i32x4 g2 = {0, 0, 0, 0};
    const i32x4 g3 = {0, 0, 0, 0};
    const i32x8 gz = {0, 0, 0, 0, 0, 0, 0, 0};   // unused trailing group
#pragma unroll
    for (int q = 0; q < 4; ++q) {
      const unsigned long long ga =
          gbase + ((unsigned long long)(q * HH + g * 32) * HH) * 2ull;
      u32x4 g0;
      g0[0] = 1u;                                      // count=1
      g0[1] = lbase + (unsigned)(q * 32 * KP * 2);     // lds_addr (bytes)
      g0[2] = (unsigned)(ga & 0xFFFFFFFFu);            // global_addr[31:0]
      g0[3] = (unsigned)((ga >> 32) & 0x01FFFFFFu)     // global_addr[56:32]
              | 0x80000000u;                           // type=2 ("image")
      // 6-arg form (clang-23 / therock-10.0 headers).
      __builtin_amdgcn_tensor_load_to_lds(g0, g1, g2, g3, gz, 0);
    }
#if __has_builtin(__builtin_amdgcn_s_wait_tensorcnt)
    __builtin_amdgcn_s_wait_tensorcnt(0);
#else
    asm volatile("s_wait_tensorcnt 0x0" ::: "memory");
#endif
  }
  __syncthreads();
#else
  // Fallback: synchronous staging with identical padded layout.
  for (int idx = tid; idx < 128 * HH; idx += 256) {
    const int lc = idx >> 10;                   // local gate-col 0..127
    const int k  = idx & (HH - 1);
    const int q  = lc >> 5;                     // gate 0..3
    const int cl = lc & 31;                     // owned col 0..31
    const int grow = q * HH + g * 32 + cl;      // W_hh row
    lds_w[lc * KP + k + (k >= 512 ? PADH : 0)] = whh_bf[(size_t)grow * HH + k];
  }
  __syncthreads();
#endif

  const int arow = rt * 16 + nloc;              // batch row for A lane
  const int hcol = g * 32 + ct * 16 + nloc;     // owned h column (B lane N)

  v8f creg = {0.f, 0.f, 0.f, 0.f, 0.f, 0.f, 0.f, 0.f};   // cell state, fp32

  for (int t = 0; t < TT; ++t) {
    const unsigned short* hrd = h_bf + (size_t)(t & 1) * BB * HH;
    unsigned short*       hwr = h_bf + (size_t)((t + 1) & 1) * BB * HH;

    if (t + 1 < TT) {   // hint next step's x_proj row into cache
      __builtin_prefetch(
          xp + ((size_t)(rt * 16 + hi * 8) * TT + (t + 1)) * G4 + hcol, 0, 2);
    }

    v8f ai = {0.f,0.f,0.f,0.f,0.f,0.f,0.f,0.f};
    v8f af = {0.f,0.f,0.f,0.f,0.f,0.f,0.f,0.f};
    v8f ag = {0.f,0.f,0.f,0.f,0.f,0.f,0.f,0.f};
    v8f ao = {0.f,0.f,0.f,0.f,0.f,0.f,0.f,0.f};

    for (int k0 = 0; k0 < HH; k0 += 32) {
      AFrag a;
      load_afrag(a, hrd + (size_t)arow * HH + k0 + hi * 16);
      const int pad   = (k0 >= 512) ? PADH : 0;
      const int bbase = ct * 16 + nloc;         // local col within gate block
      const int boff  = k0 + hi * 16 + pad;
      BFrag b0, b1, b2, b3;
      {
        const uint4* p = (const uint4*)(lds_w + (0 * 32 + bbase) * KP + boff);
        b0.q[0] = p[0]; b0.q[1] = p[1];
      }
      {
        const uint4* p = (const uint4*)(lds_w + (1 * 32 + bbase) * KP + boff);
        b1.q[0] = p[0]; b1.q[1] = p[1];
      }
      {
        const uint4* p = (const uint4*)(lds_w + (2 * 32 + bbase) * KP + boff);
        b2.q[0] = p[0]; b2.q[1] = p[1];
      }
      {
        const uint4* p = (const uint4*)(lds_w + (3 * 32 + bbase) * KP + boff);
        b3.q[0] = p[0]; b3.q[1] = p[1];
      }
      ai = __builtin_amdgcn_wmma_f32_16x16x32_bf16(false, a.v, false, b0.v,
                                                   (short)0, ai, false, false);
      af = __builtin_amdgcn_wmma_f32_16x16x32_bf16(false, a.v, false, b1.v,
                                                   (short)0, af, false, false);
      ag = __builtin_amdgcn_wmma_f32_16x16x32_bf16(false, a.v, false, b2.v,
                                                   (short)0, ag, false, false);
      ao = __builtin_amdgcn_wmma_f32_16x16x32_bf16(false, a.v, false, b3.v,
                                                   (short)0, ao, false, false);
    }

    // Gate fusion: C/D layout gives this lane (M = v + hi*8, N = nloc) for
    // all four gate tiles simultaneously -> pure register math, c in VGPRs.
#pragma unroll
    for (int v = 0; v < 8; ++v) {
      const int brow = rt * 16 + v + hi * 8;    // batch index
      const float* xr = xp + ((size_t)brow * TT + t) * G4;
      const float pi = ai[v] + xr[0 * HH + hcol];
      const float pf = af[v] + xr[1 * HH + hcol];
      const float pg = ag[v] + xr[2 * HH + hcol];
      const float po = ao[v] + xr[3 * HH + hcol];
      const float it = sigmoidf_(pi);
      const float ft = sigmoidf_(pf);
      const float gt = tanhf(pg);
      const float ot = sigmoidf_(po);
      const float cv = ft * creg[v] + it * gt;
      const float hv = ot * tanhf(cv);
      creg[v] = cv;
      out[((size_t)brow * TT + t) * HH + hcol] = hv;
      hwr[(size_t)brow * HH + hcol] = f2bf(hv);
    }

    if (t + 1 < TT) grid_barrier(bar, (unsigned)((t + 1) * NWG));
  }
}

// ---------------------------------------------------------------------------
// Workspace layout (bytes):
//   [0)      xp      : B*T*4096 fp32   = 536,870,912
//   [+)      x_bf    : B*T*256  bf16   =  16,777,216
//   [+)      wih_bf  : 4096*256 bf16   =   2,097,152
//   [+)      whh_bf  : 4096*1024 bf16  =   8,388,608
//   [+)      h_bf    : 2*64*1024 bf16  =     262,144
//   [+)      barrier : 256              (total ~ 538.3 MiB)
// ---------------------------------------------------------------------------
extern "C" void kernel_launch(void* const* d_in, const int* in_sizes, int n_in,
                              void* d_out, int out_size, void* d_ws,
                              size_t ws_size, hipStream_t stream) {
  (void)in_sizes; (void)n_in; (void)out_size; (void)ws_size;
  const float* x    = (const float*)d_in[0];   // (B,T,D)
  const float* wih  = (const float*)d_in[1];   // (4H,D)
  const float* whh  = (const float*)d_in[2];   // (4H,H)
  const float* bias = (const float*)d_in[3];   // (4H)
  float* out = (float*)d_out;                  // (B,T,H)

  char* ws = (char*)d_ws;
  const size_t off_xp    = 0;
  const size_t off_xbf   = off_xp   + (size_t)BB * TT * G4 * sizeof(float);
  const size_t off_wibf  = off_xbf  + (size_t)BB * TT * DD * 2;
  const size_t off_whbf  = off_wibf + (size_t)G4 * DD * 2;
  const size_t off_hbf   = off_whbf + (size_t)G4 * HH * 2;
  const size_t off_bar   = off_hbf  + (size_t)2 * BB * HH * 2;

  float*          xp     = (float*)(ws + off_xp);
  unsigned short* x_bf   = (unsigned short*)(ws + off_xbf);
  unsigned short* wih_bf = (unsigned short*)(ws + off_wibf);
  unsigned short* whh_bf = (unsigned short*)(ws + off_whbf);
  unsigned short* h_bf   = (unsigned short*)(ws + off_hbf);
  unsigned*       bar    = (unsigned*)(ws + off_bar);

  lstm_prep_kernel<<<8192, 256, 0, stream>>>(x, wih, whh, x_bf, wih_bf,
                                             whh_bf, h_bf, bar);
  lstm_xproj_kernel<<<65536, 256, 0, stream>>>(x_bf, wih_bf, bias, xp);
  lstm_recur_kernel<<<NWG, 256, 128 * KP * sizeof(unsigned short), stream>>>(
      xp, whh_bf, h_bf, out, bar);
}
